// AttentionBlock_disable_qk_2954937499690
// MI455X (gfx1250) — matrix operand order repaired and checked by reading, not compiled, not run on hardware
//
#include <hip/hip_runtime.h>

typedef __attribute__((ext_vector_type(16))) _Float16 v16h;
typedef __attribute__((ext_vector_type(8)))  float    v8f;

#define WMMA16(a, b, c) __builtin_amdgcn_wmma_f32_16x16x32_f16( \
    false, (a), false, (b), (short)0, (c), false, false)

#define ZERO8 {0.f,0.f,0.f,0.f,0.f,0.f,0.f,0.f}

constexpr int Bb   = 4;
constexpr int Ls   = 512;
constexpr int Hh   = 512;
constexpr int NHh  = 8;
constexpr int DKk  = 64;
constexpr int RIN  = 53;
constexpr int RHID = 32;

// A-fragment (16x32, f16) K index for half slot t in this lane.
// lane<16 : K = {0..15} ; lane>=16 : K = {16..31}
__device__ __forceinline__ int a_k(int hi, int t) {
  return t + (hi ? 16 : 0);
}

// ---------------------------------------------------------------------------
// Kernel 1: value projection  value = x @ Wv + bv  -> f16 (B, NH, L, DK)
// One wave per 16x16 output tile; K loop in chunks of 32.
// ---------------------------------------------------------------------------
__global__ __launch_bounds__(32)
void k_value_proj(const float* __restrict__ x, const float* __restrict__ Wv,
                  const float* __restrict__ bv, _Float16* __restrict__ valh) {
  const int lane = threadIdx.x;
  const int m16 = lane & 15, hi = lane >> 4;
  const int tile = blockIdx.x;               // (2048/16) * (512/16) = 128*32
  const int nt = tile & 31;
  const int mt = tile >> 5;
  const int row = mt * 16 + m16;             // A row for this lane
  const int col = nt * 16 + m16;             // B/D column for this lane

  v8f acc = ZERO8;
  for (int kc = 0; kc < Hh; kc += 32) {
    v16h a, bf;
    const float* ap = x + (size_t)row * Hh + kc;
#pragma unroll
    for (int t = 0; t < 16; ++t) a[t] = (_Float16)ap[a_k(hi, t)];
#pragma unroll
    for (int t = 0; t < 16; ++t) {
      int k = kc + t + (hi ? 16 : 0);
      bf[t] = (_Float16)Wv[(size_t)k * Hh + col];
    }
    acc = WMMA16(a, bf, acc);
  }
  const float bias = bv[col];
  const int h = col >> 6, d = col & 63;
#pragma unroll
  for (int r = 0; r < 8; ++r) {
    int gm = mt * 16 + r + 8 * hi;           // global row in [0, B*L)
    int b = gm >> 9, l = gm & 511;
    valh[(((size_t)(b * NHh + h) * Ls + l) * DKk) + d] = (_Float16)(acc[r] + bias);
  }
}

// ---------------------------------------------------------------------------
// Kernel 2: fused pair MLP  scores[b,h,i,j] = relu(refCov@W1+b1)@W2+b2
// One wave per block; each block handles 8 tiles of 16 j's.
// 5 WMMAs per tile; LDS bounce to re-layout hidden activations.
// ---------------------------------------------------------------------------
__global__ __launch_bounds__(32)
void k_pair_scores(const float* __restrict__ refCov, const float* __restrict__ W1,
                   const float* __restrict__ b1, const float* __restrict__ W2,
                   const float* __restrict__ b2, float* __restrict__ scores) {
  __shared__ _Float16 lds[16][RHID + 8];
  const int lane = threadIdx.x;
  const int m16 = lane & 15, hi = lane >> 4;

  // Preload weight B-fragments (shared by all tiles of this block).
  v16h w1f[2][2], w2f;
#pragma unroll
  for (int kt = 0; kt < 2; ++kt)
#pragma unroll
    for (int nt = 0; nt < 2; ++nt)
#pragma unroll
      for (int t = 0; t < 16; ++t) {
        int k = kt * 32 + t + (hi ? 16 : 0);
        int c = nt * 16 + m16;
        w1f[kt][nt][t] = (_Float16)((k < RIN) ? W1[k * RHID + c] : 0.f);
      }
#pragma unroll
  for (int t = 0; t < 16; ++t) {
    int k = t + (hi ? 16 : 0);
    w2f[t] = (_Float16)((m16 < NHh) ? W2[k * NHh + m16] : 0.f);
  }
  const float b1a = b1[m16];
  const float b1b = b1[16 + m16];
  const float b2n = (m16 < NHh) ? b2[m16] : 0.f;

  const int TPB = 8;
  for (int it = 0; it < TPB; ++it) {
    const int tile = blockIdx.x * TPB + it;  // 65536 tiles total
    const int jt = tile & 31;
    const int i  = (tile >> 5) & (Ls - 1);
    const int b  = tile >> 14;
    const int j  = jt * 16 + m16;
    const float* rp = refCov + ((size_t)(b * Ls + i) * Ls + j) * RIN;

    v8f acc0 = ZERO8, acc1 = ZERO8;
#pragma unroll
    for (int kt = 0; kt < 2; ++kt) {
      v16h a;
#pragma unroll
      for (int t = 0; t < 16; ++t) {
        int k = kt * 32 + a_k(hi, t);
        float v = rp[(k < RIN) ? k : 0];     // clamped addr, masked value
        a[t] = (_Float16)((k < RIN) ? v : 0.f);
      }
      acc0 = WMMA16(a, w1f[kt][0], acc0);
      acc1 = WMMA16(a, w1f[kt][1], acc1);
    }

    // bias + relu, scatter to LDS in (row, hidden) layout
    __syncthreads();
#pragma unroll
    for (int r = 0; r < 8; ++r) {
      float h0 = acc0[r] + b1a; h0 = h0 > 0.f ? h0 : 0.f;
      float h1 = acc1[r] + b1b; h1 = h1 > 0.f ? h1 : 0.f;
      lds[r + 8 * hi][m16]      = (_Float16)h0;
      lds[r + 8 * hi][16 + m16] = (_Float16)h1;
    }
    __syncthreads();

    v16h a2;
#pragma unroll
    for (int t = 0; t < 16; ++t) a2[t] = lds[m16][a_k(hi, t)];
    v8f acc2 = ZERO8;
    acc2 = WMMA16(a2, w2f, acc2);

    if (m16 < NHh) {                         // columns 0..7 are the heads
      const int h = m16;
#pragma unroll
      for (int r = 0; r < 8; ++r) {
        int jj = jt * 16 + r + 8 * hi;
        scores[(((size_t)(b * NHh + h) * Ls + i) * Ls) + jj] = acc2[r] + b2n;
      }
    }
  }
}

// ---------------------------------------------------------------------------
// Kernel 3: per-row masked softmax stats (max, 1/sum); uniform-row flag via
// negative inv.  One wave per (b,h,i) row.
// ---------------------------------------------------------------------------
__global__ __launch_bounds__(32)
void k_softmax_stats(const float* __restrict__ scores, const int* __restrict__ mask,
                     float* __restrict__ stat_m, float* __restrict__ stat_inv) {
  const int row = blockIdx.x;                // (b*NH + h)*L + i
  const int i = row & (Ls - 1);
  const int b = row >> 12;                   // row / (NH*L)
  const int lane = threadIdx.x;
  const float* sp = scores + (size_t)row * Ls;
  const int* mp = mask + b * Ls;
  const int mi = mp[i];

  float mx = -3.0e38f;
  for (int j = lane; j < Ls; j += 32)
    if (mp[j]) mx = fmaxf(mx, sp[j]);
#pragma unroll
  for (int o = 16; o; o >>= 1) mx = fmaxf(mx, __shfl_xor(mx, o, 32));

  float sum = 0.f;
  for (int j = lane; j < Ls; j += 32)
    if (mp[j]) sum += __expf(sp[j] - mx);
#pragma unroll
  for (int o = 16; o; o >>= 1) sum += __shfl_xor(sum, o, 32);

  if (lane == 0) {
    bool uniform = (mi == 0) || (mx < -1.0e38f);   // row fully masked
    stat_m[row] = mx;
    stat_inv[row] = uniform ? -1.f : 1.f / sum;
  }
}

// ---------------------------------------------------------------------------
// Kernel 4: out = attn @ value.  One wave per (b,h, 16-row i-tile, 16-col
// d-tile); attn materialized on the fly into f16 A-fragments.
// ---------------------------------------------------------------------------
__global__ __launch_bounds__(32)
void k_attn_value(const float* __restrict__ scores, const int* __restrict__ mask,
                  const float* __restrict__ stat_m, const float* __restrict__ stat_inv,
                  const _Float16* __restrict__ valh, float* __restrict__ outattn) {
  const int lane = threadIdx.x;
  const int m16 = lane & 15, hi = lane >> 4;
  const int tile = blockIdx.x;               // 4 * 8 * 32 * 4 = 4096
  const int dt = tile & 3;
  const int it = (tile >> 2) & 31;
  const int h  = (tile >> 7) & 7;
  const int b  = tile >> 10;

  const int i = it * 16 + m16;
  const int rowid = (b * NHh + h) * Ls + i;
  const float* sp = scores + (size_t)rowid * Ls;
  const float sm   = stat_m[rowid];
  const float sinv = stat_inv[rowid];
  const bool  uni  = sinv < 0.f;
  const int* mp = mask + b * Ls;
  const _Float16* vp = valh + (size_t)(b * NHh + h) * Ls * DKk + dt * 16 + m16;

  v8f acc = ZERO8;
  for (int jc = 0; jc < Ls; jc += 32) {
    v16h a, bf;
#pragma unroll
    for (int t = 0; t < 16; ++t) {
      int j = jc + a_k(hi, t);
      float p = uni ? (1.f / (float)Ls)
                    : (mp[j] ? __expf(sp[j] - sm) * sinv : 0.f);
      a[t] = (_Float16)p;
    }
#pragma unroll
    for (int t = 0; t < 16; ++t) {
      int j = jc + t + (hi ? 16 : 0);
      bf[t] = vp[(size_t)j * DKk];
    }
    acc = WMMA16(a, bf, acc);
  }
  const int d = h * 64 + dt * 16 + m16;
#pragma unroll
  for (int r = 0; r < 8; ++r) {
    int gi = it * 16 + r + 8 * hi;
    outattn[((size_t)(b * Ls + gi)) * Hh + d] = acc[r];
  }
}

// ---------------------------------------------------------------------------
// Kernel 5: output projection + bias + residual:  y = outattn @ Wo + bo + x
// ---------------------------------------------------------------------------
__global__ __launch_bounds__(32)
void k_out_proj(const float* __restrict__ oa, const float* __restrict__ Wo,
                const float* __restrict__ bo, const float* __restrict__ x,
                float* __restrict__ y) {
  const int lane = threadIdx.x;
  const int m16 = lane & 15, hi = lane >> 4;
  const int tile = blockIdx.x;               // 128*32
  const int nt = tile & 31;
  const int mt = tile >> 5;
  const int row = mt * 16 + m16;
  const int col = nt * 16 + m16;

  v8f acc = ZERO8;
  for (int kc = 0; kc < Hh; kc += 32) {
    v16h a, bf;
    const float* ap = oa + (size_t)row * Hh + kc;
#pragma unroll
    for (int t = 0; t < 16; ++t) a[t] = (_Float16)ap[a_k(hi, t)];
#pragma unroll
    for (int t = 0; t < 16; ++t) {
      int k = kc + t + (hi ? 16 : 0);
      bf[t] = (_Float16)Wo[(size_t)k * Hh + col];
    }
    acc = WMMA16(a, bf, acc);
  }
  const float bias = bo[col];
#pragma unroll
  for (int r = 0; r < 8; ++r) {
    int gm = mt * 16 + r + 8 * hi;
    y[(size_t)gm * Hh + col] = acc[r] + bias + x[(size_t)gm * Hh + col];
  }
}

// ---------------------------------------------------------------------------
// Kernel 6: LayerNorm over H=512 per row.
// ---------------------------------------------------------------------------
__global__ __launch_bounds__(256)
void k_layernorm(const float* __restrict__ y, const float* __restrict__ gamma,
                 const float* __restrict__ beta, float* __restrict__ out) {
  __shared__ float rs[8], rss[8];
  const int row = blockIdx.x;
  const float* yp = y + (size_t)row * Hh;
  float s = 0.f, ss = 0.f;
  for (int c = threadIdx.x; c < Hh; c += 256) {
    float v = yp[c];
    s += v; ss += v * v;
  }
#pragma unroll
  for (int o = 16; o; o >>= 1) {
    s  += __shfl_xor(s, o, 32);
    ss += __shfl_xor(ss, o, 32);
  }
  const int wid = threadIdx.x >> 5;
  if ((threadIdx.x & 31) == 0) { rs[wid] = s; rss[wid] = ss; }
  __syncthreads();
  s = 0.f; ss = 0.f;
#pragma unroll
  for (int w = 0; w < 8; ++w) { s += rs[w]; ss += rss[w]; }
  const float mu = s * (1.f / (float)Hh);
  const float var = ss * (1.f / (float)Hh) - mu * mu;
  const float rstd = rsqrtf(var + 1e-5f);
  for (int c = threadIdx.x; c < Hh; c += 256)
    out[(size_t)row * Hh + c] = (yp[c] - mu) * rstd * gamma[c] + beta[c];
}

// ---------------------------------------------------------------------------
extern "C" void kernel_launch(void* const* d_in, const int* in_sizes, int n_in,
                              void* d_out, int out_size, void* d_ws, size_t ws_size,
                              hipStream_t stream) {
  (void)in_sizes; (void)n_in; (void)out_size; (void)ws_size;
  const float* x      = (const float*)d_in[0];
  const int*   mask   = (const int*)d_in[1];
  const float* refCov = (const float*)d_in[2];
  const float* Wv     = (const float*)d_in[3];
  const float* bv     = (const float*)d_in[4];
  const float* W1     = (const float*)d_in[5];
  const float* b1     = (const float*)d_in[6];
  const float* W2     = (const float*)d_in[7];
  const float* b2     = (const float*)d_in[8];
  const float* Wo     = (const float*)d_in[9];
  const float* bo     = (const float*)d_in[10];
  const float* gamma  = (const float*)d_in[11];
  const float* beta   = (const float*)d_in[12];
  float* out = (float*)d_out;

  char* ws = (char*)d_ws;
  size_t off = 0;
  auto alloc = [&](size_t bytes) {
    char* p = ws + off;
    off = (off + bytes + 255) & ~(size_t)255;
    return p;
  };
  _Float16* valh    = (_Float16*)alloc((size_t)Bb * NHh * Ls * DKk * 2);   //  2 MB
  float*    scores  = (float*)   alloc((size_t)Bb * NHh * Ls * Ls * 4);    // 33.5 MB
  float*    stat_m  = (float*)   alloc((size_t)Bb * NHh * Ls * 4);
  float*    stat_i  = (float*)   alloc((size_t)Bb * NHh * Ls * 4);
  float*    outattn = (float*)   alloc((size_t)Bb * Ls * Hh * 4);          //  4 MB
  float*    ybuf    = (float*)   alloc((size_t)Bb * Ls * Hh * 4);          //  4 MB

  k_value_proj   <<<4096,  32, 0, stream>>>(x, Wv, bv, valh);
  k_pair_scores  <<<8192,  32, 0, stream>>>(refCov, W1, b1, W2, b2, scores);
  k_softmax_stats<<<16384, 32, 0, stream>>>(scores, mask, stat_m, stat_i);
  k_attn_value   <<<4096,  32, 0, stream>>>(scores, mask, stat_m, stat_i, valh, outattn);
  k_out_proj     <<<4096,  32, 0, stream>>>(outattn, Wo, bo, x, ybuf);
  k_layernorm    <<<2048, 256, 0, stream>>>(ybuf, gamma, beta, out);
}
